// LandmarkEncoder_18090402251004
// MI455X (gfx1250) — matrix operand order repaired and checked by reading, not compiled, not run on hardware
//
#include <hip/hip_runtime.h>
#include <hip/hip_bf16.h>

// ---------------------------------------------------------------------------
// LandmarkEncoder for gfx1250 (MI455X). Heavy GEMMs on v_wmma_f32_16x16x32_f16
// (f16 in / f32 accumulate); LDS staging via global_load_async_to_lds_b128
// (ASYNCcnt path) instead of the VGPR round-trip.
// ---------------------------------------------------------------------------

typedef __attribute__((ext_vector_type(16))) _Float16 v16h;
typedef __attribute__((ext_vector_type(8)))  _Float16 v8h;
typedef __attribute__((ext_vector_type(8)))  float    v8f;

union U16 { v16h v; v8h h[2]; };

#define WMMA(a, b, c) \
  __builtin_amdgcn_wmma_f32_16x16x32_f16(false, (a), false, (b), (short)0, (c), false, false)

#define NB   8      // batch
#define NK   68     // landmarks / nodes
#define NT   512    // input time
#define NT2  256    // time after stride-2 conv
#define NC   128    // channels
#define NHD  4      // heads
#define HC   32     // head channels
#define NFF  512    // ffn hidden
#define KS5  5      // conv kernel taps

// Direct global->LDS async copy (16B per active lane). The LDS operand is the
// wave-relative byte offset = low 32 bits of the generic pointer (flat LDS
// aperture truncation). Tracked by ASYNCcnt.
__device__ __forceinline__ void async_ld_b128(void* lds_dst, const void* gsrc) {
  unsigned loff = (unsigned)(uintptr_t)lds_dst;
  unsigned long long ga = (unsigned long long)(uintptr_t)gsrc;
  asm volatile("global_load_async_to_lds_b128 %0, %1, off"
               :: "v"(loff), "v"(ga) : "memory");
}
__device__ __forceinline__ void wait_async() {
  asm volatile("s_wait_asynccnt 0" ::: "memory");
}

// ============================================================================
// Kernel 0: repack weights to f16 B^T layouts + fold BatchNorms
// ============================================================================
__global__ __launch_bounds__(256) void repack_k(
    const float* __restrict__ conv_w,                      // (2,C,C,1,5)
    const float* __restrict__ qw, const float* __restrict__ kw, const float* __restrict__ vw, // (L,32,32)
    const float* __restrict__ w1, const float* __restrict__ w2,  // (L,128,512),(L,512,128)
    const float* __restrict__ cg, const float* __restrict__ cb,
    const float* __restrict__ cm, const float* __restrict__ cv,  // (3,C)
    const float* __restrict__ g1, const float* __restrict__ b1,
    const float* __restrict__ m1, const float* __restrict__ v1,  // (L,C)
    const float* __restrict__ g2, const float* __restrict__ b2,
    const float* __restrict__ m2, const float* __restrict__ v2,  // (L,C)
    _Float16* __restrict__ Wc,  _Float16* __restrict__ Wq,
    _Float16* __restrict__ Wk,  _Float16* __restrict__ Wv,
    _Float16* __restrict__ W1t, _Float16* __restrict__ W2t,
    float* __restrict__ cbns, float* __restrict__ cbnt,
    float* __restrict__ bn1s, float* __restrict__ bn1t,
    float* __restrict__ bn2s, float* __restrict__ bn2t)
{
  int i = blockIdx.x * 256 + threadIdx.x;
  // Wc: [l][co][kt*128+ci] <- conv_w[l][co][ci][0][kt]   (2*128*640 = 163840)
  if (i < 163840) {
    int ci = i & 127, kt = (i >> 7) % 5, co = (i / 640) % 128, l = i / 81920;
    Wc[i] = (_Float16)conv_w[((l * 128 + co) * 128 + ci) * 5 + kt];
    return;
  }
  i -= 163840;
  // Wq/Wk/Wv: [l][e][d] <- w[l][d][e]  (3 * 2048)
  if (i < 6144) {
    int which = i / 2048, j = i % 2048;
    int d = j & 31, e = (j >> 5) & 31, l = j >> 10;
    const float* src = which == 0 ? qw : (which == 1 ? kw : vw);
    _Float16*   dst = which == 0 ? Wq : (which == 1 ? Wk : Wv);
    dst[j] = (_Float16)src[l * 1024 + d * 32 + e];
    return;
  }
  i -= 6144;
  // W1t: [l][e<512][d<128] <- w1[l][d][e]  (131072)
  if (i < 131072) {
    int d = i & 127, e = (i >> 7) & 511, l = i >> 16;
    W1t[i] = (_Float16)w1[(l * 128 + d) * 512 + e];
    return;
  }
  i -= 131072;
  // W2t: [l][c<128][e<512] <- w2[l][e][c]  (131072)
  if (i < 131072) {
    int e = i & 511, c = (i >> 9) & 127, l = i >> 16;
    W2t[i] = (_Float16)w2[(l * 512 + e) * 128 + c];
    return;
  }
  i -= 131072;
  if (i < 384) { float s = cg[i] * rsqrtf(cv[i] + 1e-5f); cbns[i] = s; cbnt[i] = cb[i] - cm[i] * s; return; }
  i -= 384;
  if (i < 256) { float s = g1[i] * rsqrtf(v1[i] + 1e-5f); bn1s[i] = s; bn1t[i] = b1[i] - m1[i] * s; return; }
  i -= 256;
  if (i < 256) { float s = g2[i] * rsqrtf(v2[i] + 1e-5f); bn2s[i] = s; bn2t[i] = b2[i] - m2[i] * s; return; }
}

// ============================================================================
// Kernel 1: conv0 (in=2, stride 2 on T) + bias + BN0 + ReLU -> h (B,K,T2,C) f16
// ============================================================================
__global__ __launch_bounds__(256) void conv0_k(
    const float* __restrict__ x,   // (B,2,K,T)
    const float* __restrict__ w0,  // (C,2,1,5)
    const float* __restrict__ b0,
    const float* __restrict__ bns, const float* __restrict__ bnt,
    _Float16* __restrict__ hout)   // (B,K,T2,C)
{
  int idx = blockIdx.x * 256 + threadIdx.x;
  if (idx >= NB * NK * NT2 * NC) return;
  int c  = idx & 127;
  int t  = (idx >> 7) & 255;
  int bk = idx >> 15;
  int k  = bk % NK, b = bk / NK;
  float acc = 0.f;
  int tin0 = 2 * t - 2;
  for (int ci = 0; ci < 2; ++ci) {
    const float* xr = x + ((b * 2 + ci) * NK + k) * NT;
    const float* wr = w0 + (c * 2 + ci) * 5;
    for (int kt = 0; kt < 5; ++kt) {
      int ti = tin0 + kt;
      if (ti >= 0 && ti < NT) acc += xr[ti] * wr[kt];
    }
  }
  float val = (acc + b0[c]) * bns[c] + bnt[c];
  hout[idx] = (_Float16)(val > 0.f ? val : 0.f);
}

// ============================================================================
// Kernel 2: conv (C->C, k=5, pad 2) as implicit GEMM via WMMA, + bias+BN+ReLU.
// One block per (b,k, 32-row T tile). 8 waves; each wave -> 16x32 output tile.
// Reduction r = kt*128 + ci over 640 (20 k-steps of 32).
// ============================================================================
__global__ __launch_bounds__(256) void conv_k(
    const _Float16* __restrict__ hin,  // (B,K,T2,C)
    const _Float16* __restrict__ Wc,   // (128,640) = B^T
    const float* __restrict__ bias, const float* __restrict__ bns,
    const float* __restrict__ bnt,
    _Float16* __restrict__ hout)       // (B,K,T2,C)
{
  __shared__ _Float16 hs[36 * 128];    // 32-row tile + 4 halo rows
  int blk = blockIdx.x;
  int tt  = (blk & 7) * 32;            // T2/32 = 8 tiles
  int bk  = blk >> 3;
  int k   = bk % NK, b = bk / NK;
  int tid = threadIdx.x;

  // Stage haloed input slab directly into LDS (async path); zero the
  // out-of-range halo rows with plain DS stores (disjoint addresses).
  for (int ch = tid; ch < 36 * 16; ch += 256) {
    int i = ch >> 4, c8 = (ch & 15) << 3;
    int t = tt + i - 2;
    _Float16* dst = hs + i * 128 + c8;
    if (t >= 0 && t < NT2) {
      async_ld_b128(dst, hin + ((size_t)(b * NK + k) * NT2 + t) * NC + c8);
    } else {
      v8h z = {}; *(v8h*)dst = z;
    }
  }
  wait_async();
  __syncthreads();

  int wave = tid >> 5, lane = tid & 31;
  int mt  = wave >> 2;                 // 0..1
  int nt0 = (wave & 3) * 2;            // 0,2,4,6
  int m   = mt * 16 + (lane & 15);
  int n0  = nt0 * 16 + (lane & 15);
  int n1  = n0 + 16;
  v8f acc0 = {}, acc1 = {};

  for (int ks = 0; ks < 20; ++ks) {
    U16 a, bf0, bf1;
    int r0 = ks * 32 + ((lane >> 4) << 3);
    int kt0 = r0 >> 7, ci0 = r0 & 127;
    a.h[0] = *(const v8h*)(hs + (m + kt0) * 128 + ci0);
    int r1 = r0 + 16;
    int kt1 = r1 >> 7, ci1 = r1 & 127;
    a.h[1] = *(const v8h*)(hs + (m + kt1) * 128 + ci1);
    int kb = ks * 32 + ((lane >> 4) << 3);
    bf0.h[0] = *(const v8h*)(Wc + n0 * 640 + kb);
    bf0.h[1] = *(const v8h*)(Wc + n0 * 640 + kb + 16);
    bf1.h[0] = *(const v8h*)(Wc + n1 * 640 + kb);
    bf1.h[1] = *(const v8h*)(Wc + n1 * 640 + kb + 16);
    acc0 = WMMA(a.v, bf0.v, acc0);
    acc1 = WMMA(a.v, bf1.v, acc1);
  }

  float s0 = bns[n0], sh0 = bnt[n0], bi0 = bias[n0];
  float s1 = bns[n1], sh1 = bnt[n1], bi1 = bias[n1];
  for (int j = 0; j < 8; ++j) {
    int ml = mt * 16 + ((lane >> 4) << 3) + j;
    int t  = tt + ml;
    size_t base = ((size_t)(b * NK + k) * NT2 + t) * NC;
    float v0 = (acc0[j] + bi0) * s0 + sh0; v0 = v0 > 0.f ? v0 : 0.f;
    float v1 = (acc1[j] + bi1) * s1 + sh1; v1 = v1 > 0.f ? v1 : 0.f;
    hout[base + n0] = (_Float16)v0;
    hout[base + n1] = (_Float16)v1;
  }
}

// ============================================================================
// Kernel 3: add semantic embedding + transpose (B,K,T2,C)->(B,T2,K,C);
// keep fp32 master copy o and f16 mirror oh.
// ============================================================================
__global__ __launch_bounds__(256) void se_k(
    const _Float16* __restrict__ h, const int* __restrict__ lidx,
    const float* __restrict__ emb,
    float* __restrict__ o, _Float16* __restrict__ oh)
{
  int idx = blockIdx.x * 256 + threadIdx.x;   // over (b,t,k,c)
  if (idx >= NB * NT2 * NK * NC) return;
  int c  = idx & 127;
  int tk = idx >> 7;
  int k  = tk % NK;
  int bt = tk / NK;
  int t  = bt & 255, b = bt >> 8;
  float val = (float)h[((size_t)(b * NK + k) * NT2 + t) * NC + c]
            + emb[lidx[b * NK + k] * NC + c];
  o[idx]  = val;
  oh[idx] = (_Float16)val;
}

// ============================================================================
// Kernel 4: fused multi-head node attention per (b,t). 8 waves.
// q/k/v proj -> S=QK^T -> softmax over 68 keys -> +a_se -> att@V
// -> +residual -> BN1. Updates o (fp32) and oh (f16) in place.
// ============================================================================
__global__ __launch_bounds__(256) void attn_k(
    const float* __restrict__ qb, const float* __restrict__ kb2,
    const float* __restrict__ vb,                            // (32) each
    const _Float16* __restrict__ Wq, const _Float16* __restrict__ Wk,
    const _Float16* __restrict__ Wv,                         // (32,32) B^T
    const float* __restrict__ a_se,                          // (68,68)
    const float* __restrict__ bns, const float* __restrict__ bnt,
    float* __restrict__ o, _Float16* __restrict__ oh)
{
  extern __shared__ float4 smem_f4[];
  _Float16* X   = (_Float16*)smem_f4;        // 80 x 128
  _Float16* Qs  = X   + 80 * 128;            // 4 x 80 x 32
  _Float16* Ksm = Qs  + 4 * 80 * 32;         // 4 x 80 x 32
  _Float16* Vt  = Ksm + 4 * 80 * 32;         // 4 x 32 x 96 (transposed V)
  _Float16* Af  = Vt  + 4 * 32 * 96;         // 80 x 96 (f16 attention)
  float*    S   = (float*)(Af + 80 * 96);    // 80 x 80

  int bt = blockIdx.x;
  size_t obase = (size_t)bt * (NK * NC);
  int tid = threadIdx.x, wave = tid >> 5, lane = tid & 31;

  for (int ch = tid; ch < 80 * 16; ch += 256) {
    int r = ch >> 4, c8 = (ch & 15) << 3;
    _Float16* dst = X + r * 128 + c8;
    if (r < NK) {
      async_ld_b128(dst, oh + obase + r * NC + c8);
    } else {
      v8h z = {}; *(v8h*)dst = z;
    }
  }
  for (int i = tid; i < (4 * 32 * 96) / 8; i += 256) { v8h z = {}; *(v8h*)(Vt + i * 8) = z; }
  for (int i = tid; i < (80 * 96) / 8;     i += 256) { v8h z = {}; *(v8h*)(Af + i * 8) = z; }
  wait_async();
  __syncthreads();

  // ---- QKV projections: 3 proj x 4 heads x 5 mtiles x 2 ntiles = 120 jobs
  for (int job = wave; job < 120; job += 8) {
    int p = job / 40, rem = job % 40;
    int hd = rem / 10, mt = (rem % 10) / 2, nt = rem % 2;
    const _Float16* W  = p == 0 ? Wq : (p == 1 ? Wk : Wv);
    const float*    bi = p == 0 ? qb : (p == 1 ? kb2 : vb);
    U16 a, bf;
    int m  = mt * 16 + (lane & 15);
    int k0 = (lane >> 4) << 3;
    a.h[0] = *(const v8h*)(X + m * 128 + hd * 32 + k0);
    a.h[1] = *(const v8h*)(X + m * 128 + hd * 32 + k0 + 16);
    int n  = nt * 16 + (lane & 15);
    int kb = (lane >> 4) << 3;
    bf.h[0] = *(const v8h*)(W + n * 32 + kb);
    bf.h[1] = *(const v8h*)(W + n * 32 + kb + 16);
    v8f acc = {};
    acc = WMMA(a.v, bf.v, acc);
    float bv = bi[n];
    if (p == 2) {            // V stored transposed: Vt[hd][e][kn]
      v8h pk;
      for (int j = 0; j < 8; ++j) pk[j] = (_Float16)(acc[j] + bv);
      int mb = mt * 16 + ((lane >> 4) << 3);
      *(v8h*)(Vt + (hd * 32 + n) * 96 + mb) = pk;
    } else {
      _Float16* dst = (p == 0 ? Qs : Ksm) + hd * 80 * 32;
      for (int j = 0; j < 8; ++j) {
        int ml = mt * 16 + ((lane >> 4) << 3) + j;
        dst[ml * 32 + n] = (_Float16)(acc[j] + bv);
      }
    }
  }
  __syncthreads();

  // ---- per head: S = Q K^T, softmax(+a_se), out = att @ V
  for (int hh = 0; hh < NHD; ++hh) {
    const _Float16* Qh = Qs  + hh * 80 * 32;
    const _Float16* Kh = Ksm + hh * 80 * 32;
    const _Float16* Vh = Vt  + hh * 32 * 96;

    for (int job = wave; job < 25; job += 8) {   // 5x5 S tiles, K=32 (one step)
      int mt = job / 5, nt = job % 5;
      U16 a, bf;
      int m  = mt * 16 + (lane & 15);
      int k0 = (lane >> 4) << 3;
      a.h[0] = *(const v8h*)(Qh + m * 32 + k0);
      a.h[1] = *(const v8h*)(Qh + m * 32 + k0 + 16);
      int n  = nt * 16 + (lane & 15);
      int kb = (lane >> 4) << 3;
      bf.h[0] = *(const v8h*)(Kh + n * 32 + kb);
      bf.h[1] = *(const v8h*)(Kh + n * 32 + kb + 16);
      v8f acc = {};
      acc = WMMA(a.v, bf.v, acc);
      for (int j = 0; j < 8; ++j) {
        int ml = mt * 16 + ((lane >> 4) << 3) + j;
        S[ml * 80 + n] = acc[j];
      }
    }
    __syncthreads();

    for (int r = tid; r < NK; r += 256) {        // softmax rows (68 valid keys)
      float mx = -1e30f;
      for (int c2 = 0; c2 < NK; ++c2) mx = fmaxf(mx, S[r * 80 + c2]);
      float sum = 0.f;
      for (int c2 = 0; c2 < NK; ++c2) { float e = __expf(S[r * 80 + c2] - mx); S[r * 80 + c2] = e; sum += e; }
      float inv = 1.f / sum;
      for (int c2 = 0; c2 < NK; ++c2)
        Af[r * 96 + c2] = (_Float16)(S[r * 80 + c2] * inv + a_se[r * NK + c2]);
    }
    __syncthreads();

    for (int job = wave; job < 10; job += 8) {   // 5x2 out tiles, K=96 (3 steps)
      int mt = job / 2, nt = job % 2;
      int m  = mt * 16 + (lane & 15);
      int n  = nt * 16 + (lane & 15);
      v8f acc = {};
      for (int ks = 0; ks < 3; ++ks) {
        U16 a, bf;
        int k0 = ks * 32 + ((lane >> 4) << 3);
        a.h[0] = *(const v8h*)(Af + m * 96 + k0);
        a.h[1] = *(const v8h*)(Af + m * 96 + k0 + 16);
        int kb = ks * 32 + ((lane >> 4) << 3);
        bf.h[0] = *(const v8h*)(Vh + n * 96 + kb);
        bf.h[1] = *(const v8h*)(Vh + n * 96 + kb + 16);
        acc = WMMA(a.v, bf.v, acc);
      }
      int c = hh * 32 + nt * 16 + (lane & 15);
      float s = bns[c], sh = bnt[c];
      for (int j = 0; j < 8; ++j) {
        int q = mt * 16 + ((lane >> 4) << 3) + j;
        if (q < NK) {
          size_t idx = obase + (size_t)q * NC + c;
          float val = (acc[j] + o[idx]) * s + sh;
          o[idx]  = val;
          oh[idx] = (_Float16)val;
        }
      }
    }
    __syncthreads();
  }
}

// ============================================================================
// Kernel 5: fused FFN per (b,t): relu(X@W1+b1)@W2 + b2 + X -> BN2 (in place).
// ============================================================================
__global__ __launch_bounds__(256) void ffn_k(
    const _Float16* __restrict__ W1t,  // (512,128) B^T
    const _Float16* __restrict__ W2t,  // (128,512) B^T
    const float* __restrict__ b1, const float* __restrict__ b2,
    const float* __restrict__ bns, const float* __restrict__ bnt,
    float* __restrict__ o, _Float16* __restrict__ oh)
{
  extern __shared__ float4 smem_f4[];
  _Float16* X   = (_Float16*)smem_f4;  // 80 x 128
  _Float16* Mid = X + 80 * 128;        // 80 x 512

  int bt = blockIdx.x;
  size_t obase = (size_t)bt * (NK * NC);
  int tid = threadIdx.x, wave = tid >> 5, lane = tid & 31;

  for (int ch = tid; ch < 80 * 16; ch += 256) {
    int r = ch >> 4, c8 = (ch & 15) << 3;
    _Float16* dst = X + r * 128 + c8;
    if (r < NK) {
      async_ld_b128(dst, oh + obase + r * NC + c8);
    } else {
      v8h z = {}; *(v8h*)dst = z;
    }
  }
  wait_async();
  __syncthreads();

  // GEMM1: 5 x 32 tiles, K = 128 (4 steps)
  for (int job = wave; job < 160; job += 8) {
    int mt = job >> 5, nt = job & 31;
    int m = mt * 16 + (lane & 15);
    int n = nt * 16 + (lane & 15);
    v8f acc = {};
    for (int ks = 0; ks < 4; ++ks) {
      U16 a, bf;
      int k0 = ks * 32 + ((lane >> 4) << 3);
      a.h[0] = *(const v8h*)(X + m * 128 + k0);
      a.h[1] = *(const v8h*)(X + m * 128 + k0 + 16);
      int kb = ks * 32 + ((lane >> 4) << 3);
      bf.h[0] = *(const v8h*)(W1t + n * 128 + kb);
      bf.h[1] = *(const v8h*)(W1t + n * 128 + kb + 16);
      acc = WMMA(a.v, bf.v, acc);
    }
    float bv = b1[n];
    for (int j = 0; j < 8; ++j) {
      int ml = mt * 16 + ((lane >> 4) << 3) + j;
      float v = acc[j] + bv;
      Mid[ml * 512 + n] = (_Float16)(v > 0.f ? v : 0.f);
    }
  }
  __syncthreads();

  // GEMM2: 5 x 8 tiles, K = 512 (16 steps)
  for (int job = wave; job < 40; job += 8) {
    int mt = job >> 3, nt = job & 7;
    int m = mt * 16 + (lane & 15);
    int n = nt * 16 + (lane & 15);
    v8f acc = {};
    for (int ks = 0; ks < 16; ++ks) {
      U16 a, bf;
      int k0 = ks * 32 + ((lane >> 4) << 3);
      a.h[0] = *(const v8h*)(Mid + m * 512 + k0);
      a.h[1] = *(const v8h*)(Mid + m * 512 + k0 + 16);
      int kb = ks * 32 + ((lane >> 4) << 3);
      bf.h[0] = *(const v8h*)(W2t + n * 512 + kb);
      bf.h[1] = *(const v8h*)(W2t + n * 512 + kb + 16);
      acc = WMMA(a.v, bf.v, acc);
    }
    float bv = b2[n], s = bns[n], sh = bnt[n];
    for (int j = 0; j < 8; ++j) {
      int q = mt * 16 + ((lane >> 4) << 3) + j;
      if (q < NK) {
        size_t idx = obase + (size_t)q * NC + n;
        float val = (acc[j] + bv + o[idx]) * s + sh;
        o[idx]  = val;
        oh[idx] = (_Float16)val;
      }
    }
  }
}

// ============================================================================
// Kernel 6: mean over K -> (B,T2,C)
// ============================================================================
__global__ __launch_bounds__(256) void mean_k(const float* __restrict__ o,
                                              float* __restrict__ out)
{
  int idx = blockIdx.x * 256 + threadIdx.x;
  if (idx >= NB * NT2 * NC) return;
  int c = idx & 127, bt = idx >> 7;
  const float* p = o + (size_t)bt * (NK * NC) + c;
  float s = 0.f;
  for (int k = 0; k < NK; ++k) s += p[k * NC];
  out[idx] = s * (1.0f / NK);
}

// ============================================================================
extern "C" void kernel_launch(void* const* d_in, const int* in_sizes, int n_in,
                              void* d_out, int out_size, void* d_ws, size_t ws_size,
                              hipStream_t stream) {
  const float* x       = (const float*)d_in[0];
  const int*   lidx    = (const int*)  d_in[1];
  const float* conv0_w = (const float*)d_in[2];
  const float* conv0_b = (const float*)d_in[3];
  const float* conv_w  = (const float*)d_in[4];
  const float* conv_b  = (const float*)d_in[5];
  const float* cbn_g   = (const float*)d_in[6];
  const float* cbn_b   = (const float*)d_in[7];
  const float* cbn_m   = (const float*)d_in[8];
  const float* cbn_v   = (const float*)d_in[9];
  const float* emb     = (const float*)d_in[10];
  const float* q_w     = (const float*)d_in[11];
  const float* q_b     = (const float*)d_in[12];
  const float* k_w     = (const float*)d_in[13];
  const float* k_b     = (const float*)d_in[14];
  const float* v_w     = (const float*)d_in[15];
  const float* v_b     = (const float*)d_in[16];
  const float* a_se    = (const float*)d_in[17];
  const float* bn1_g   = (const float*)d_in[18];
  const float* bn1_b   = (const float*)d_in[19];
  const float* bn1_m   = (const float*)d_in[20];
  const float* bn1_v   = (const float*)d_in[21];
  const float* ffn_w1  = (const float*)d_in[22];
  const float* ffn_b1  = (const float*)d_in[23];
  const float* ffn_w2  = (const float*)d_in[24];
  const float* ffn_b2  = (const float*)d_in[25];
  const float* bn2_g   = (const float*)d_in[26];
  const float* bn2_b   = (const float*)d_in[27];
  const float* bn2_m   = (const float*)d_in[28];
  const float* bn2_v   = (const float*)d_in[29];

  char* ws = (char*)d_ws;
  size_t off = 0;
  auto carve = [&](size_t bytes) -> void* {
    off = (off + 255) & ~(size_t)255;
    void* p = ws + off;
    off += bytes;
    return p;
  };

  const size_t ACT = (size_t)NB * NK * NT2 * NC;   // 17,825,792 elements

  _Float16* Wc   = (_Float16*)carve(163840 * 2);
  _Float16* Wq   = (_Float16*)carve(2048 * 2);
  _Float16* Wk   = (_Float16*)carve(2048 * 2);
  _Float16* Wv   = (_Float16*)carve(2048 * 2);
  _Float16* W1t  = (_Float16*)carve(131072 * 2);
  _Float16* W2t  = (_Float16*)carve(131072 * 2);
  float*    cbns = (float*)carve(384 * 4);
  float*    cbnt = (float*)carve(384 * 4);
  float*    bn1s = (float*)carve(256 * 4);
  float*    bn1t = (float*)carve(256 * 4);
  float*    bn2s = (float*)carve(256 * 4);
  float*    bn2t = (float*)carve(256 * 4);
  _Float16* h0   = (_Float16*)carve(ACT * 2);
  _Float16* h1   = (_Float16*)carve(ACT * 2);
  float*    o    = (float*)carve(ACT * 4);
  _Float16* oh   = (_Float16*)carve(ACT * 2);

  // 0: weight repack + BN folds
  repack_k<<<1692, 256, 0, stream>>>(conv_w, q_w, k_w, v_w, ffn_w1, ffn_w2,
      cbn_g, cbn_b, cbn_m, cbn_v, bn1_g, bn1_b, bn1_m, bn1_v,
      bn2_g, bn2_b, bn2_m, bn2_v,
      Wc, Wq, Wk, Wv, W1t, W2t, cbns, cbnt, bn1s, bn1t, bn2s, bn2t);

  // 1: conv0 + BN0 + ReLU
  conv0_k<<<(int)(ACT / 256), 256, 0, stream>>>(x, conv0_w, conv0_b,
      cbns, cbnt, h0);

  // 2: conv1, conv2 (WMMA implicit GEMM)
  conv_k<<<NB * NK * (NT2 / 32), 256, 0, stream>>>(h0, Wc,
      conv_b, cbns + 128, cbnt + 128, h1);
  conv_k<<<NB * NK * (NT2 / 32), 256, 0, stream>>>(h1, Wc + 81920,
      conv_b + 128, cbns + 256, cbnt + 256, h0);

  // 3: + semantic embedding, transpose to (B,T2,K,C)
  se_k<<<(int)(ACT / 256), 256, 0, stream>>>(h0, lidx, emb, o, oh);

  // 4: L transformer layers (attention + FFN), fused per (b,t)
  const int attn_lds = (80 * 128 + 2 * 4 * 80 * 32 + 4 * 32 * 96 + 80 * 96) * 2
                     + 80 * 80 * 4;                       // 126,976 B
  const int ffn_lds  = (80 * 128 + 80 * 512) * 2;         // 102,400 B
  for (int l = 0; l < 2; ++l) {
    attn_k<<<NB * NT2, 256, attn_lds, stream>>>(
        q_b + l * 32, k_b + l * 32, v_b + l * 32,
        Wq + l * 1024, Wk + l * 1024, Wv + l * 1024,
        a_se + l * NK * NK, bn1s + l * 128, bn1t + l * 128, o, oh);
    ffn_k<<<NB * NT2, 256, ffn_lds, stream>>>(
        W1t + l * 65536, W2t + l * 65536,
        ffn_b1 + l * 512, ffn_b2 + l * 128,
        bn2s + l * 128, bn2t + l * 128, o, oh);
  }

  // 5: mean over K
  mean_k<<<(NB * NT2 * NC) / 256, 256, 0, stream>>>(o, (float*)d_out);
}
